// LlamaDecoderLayerDAT_33835752358188
// MI455X (gfx1250) — matrix operand order repaired and checked by reading, not compiled, not run on hardware
//
#include <hip/hip_runtime.h>
#include <hip/hip_bf16.h>

// ---------------- problem constants ----------------
#define B_     2
#define NQ_    1024
#define C_     2048
#define NH_    16
#define HD_    128
#define OFFG_  4
#define OFFD_  512     // C_/OFFG_
#define INTER_ 256
#define LR_    24
#define HR_    48
#define NIMG_  576     // LR_*LR_
#define DFF_   8192
#define KL_    1600    // NQ_ + NIMG_
#define BG_    8       // B_*OFFG_
#define ROPE_THETA_ 10000.0f

// ---------------- types ----------------
typedef __attribute__((ext_vector_type(16))) __bf16 bf16x16;
typedef __attribute__((ext_vector_type(8)))  float  f32x8;

union AFrag { bf16x16 v; uint4 u[2]; };

__device__ __forceinline__ unsigned short f32_to_bf16_bits(float f) {
  union { float f; unsigned u; } a; a.f = f;
  unsigned r = a.u + 0x7FFFu + ((a.u >> 16) & 1u);
  return (unsigned short)(r >> 16);
}

// =====================================================================
// Generic bf16 WMMA GEMM:  C[M,N] = A[M,K] * B[K,N] (+bias) (+resid)
// block = 256 threads (8 waves); tile 128x128; K-step 32
// =====================================================================
#define TM 128
#define TN 128
#define TK 32

__global__ __launch_bounds__(256)
void gemm_bf16_wmma(const unsigned short* __restrict__ A,
                    const unsigned short* __restrict__ Bm,
                    const float* __restrict__ bias,
                    const float* __restrict__ resid,
                    float* __restrict__ outF,
                    unsigned short* __restrict__ outB,
                    int M, int N, int K)
{
  __shared__ unsigned short As[TM][TK + 8];   // row-major (m, k)
  __shared__ unsigned short Bt[TN][TK + 8];   // transposed (n, k)

  const int tid   = threadIdx.x;
  const int wave  = tid >> 5;
  const int lane  = tid & 31;
  const int lhalf = lane >> 4;
  const int l16   = lane & 15;

  const int m0 = blockIdx.y * TM;
  const int n0 = blockIdx.x * TN;
  const int wm = (wave & 3) * 32;   // 4 waves along M
  const int wn = (wave >> 2) * 64;  // 2 waves along N

  f32x8 acc[2][4];
  #pragma unroll
  for (int f = 0; f < 2; ++f)
    #pragma unroll
    for (int g = 0; g < 4; ++g)
      #pragma unroll
      for (int e = 0; e < 8; ++e) acc[f][g][e] = 0.f;

  for (int k0 = 0; k0 < K; k0 += TK) {
    { // stage A tile 128x32 (coalesced)
      int row = tid >> 1;
      int col = (tid & 1) * 16;
      const unsigned short* src = A + (size_t)(m0 + row) * K + k0 + col;
      uint4 a0 = *(const uint4*)(src);
      uint4 a1 = *(const uint4*)(src + 8);
      *(uint4*)&As[row][col]     = a0;
      *(uint4*)&As[row][col + 8] = a1;
    }
    { // stage B tile 32x128, transposed into Bt
      int row = tid >> 3;            // k
      int col = (tid & 7) * 16;      // n
      const unsigned short* src = Bm + (size_t)(k0 + row) * N + n0 + col;
      unsigned short tmp[16];
      *(uint4*)&tmp[0] = *(const uint4*)(src);
      *(uint4*)&tmp[8] = *(const uint4*)(src + 8);
      #pragma unroll
      for (int e = 0; e < 16; ++e) Bt[col + e][row] = tmp[e];
    }
    if (k0 + TK < K) {
      __builtin_prefetch(A  + (size_t)(m0 + (tid >> 1)) * K + (k0 + TK) + (tid & 1) * 16, 0, 1);
      __builtin_prefetch(Bm + (size_t)(k0 + TK + (tid >> 3)) * N + n0 + (tid & 7) * 16, 0, 1);
    }
    __syncthreads();

    AFrag af[2];
    #pragma unroll
    for (int f = 0; f < 2; ++f) {
      int row = wm + f * 16 + l16;
      int ko  = lhalf * 8;
      af[f].u[0] = *(const uint4*)&As[row][ko];
      af[f].u[1] = *(const uint4*)&As[row][16 + ko];
    }
    #pragma unroll
    for (int g = 0; g < 4; ++g) {
      AFrag bf_;
      int nrow = wn + g * 16 + l16;
      int ko   = lhalf * 8;
      bf_.u[0] = *(const uint4*)&Bt[nrow][ko];
      bf_.u[1] = *(const uint4*)&Bt[nrow][16 + ko];
      #pragma unroll
      for (int f = 0; f < 2; ++f) {
        acc[f][g] = __builtin_amdgcn_wmma_f32_16x16x32_bf16(
            false, af[f].v, false, bf_.v, (short)0, acc[f][g], false, false);
      }
    }
    __syncthreads();
  }

  #pragma unroll
  for (int f = 0; f < 2; ++f)
    #pragma unroll
    for (int g = 0; g < 4; ++g)
      #pragma unroll
      for (int r = 0; r < 8; ++r) {
        int gm = m0 + wm + f * 16 + lhalf * 8 + r;
        int gn = n0 + wn + g * 16 + l16;
        float v = acc[f][g][r];
        if (bias)  v += bias[gn];
        if (resid) v += resid[(size_t)gm * N + gn];
        if (outF)  outF[(size_t)gm * N + gn] = v;
        if (outB)  outB[(size_t)gm * N + gn] = f32_to_bf16_bits(v);
      }
}

// =====================================================================
// Flash attention with WMMA.  grid = (NQ/128, B*NH), block = 256.
// Q:(B,NH,NQ,HD)bf16  K,V:(B,NH,KL,HD)bf16  O:(B,NQ,C)bf16
// =====================================================================
__global__ __launch_bounds__(256)
void attn_flash_wmma(const unsigned short* __restrict__ Qb,
                     const unsigned short* __restrict__ Kb,
                     const unsigned short* __restrict__ Vb,
                     unsigned short* __restrict__ Ob,
                     float scale)
{
  __shared__ unsigned short Ks[32][HD_ + 8];     // (key, hd)
  __shared__ unsigned short Vt[HD_][32 + 8];     // (hd, key) transposed
  __shared__ unsigned short Ps[8][16][32 + 8];   // per-wave P tile

  const int tid   = threadIdx.x;
  const int wave  = tid >> 5;
  const int lane  = tid & 31;
  const int lhalf = lane >> 4;
  const int l16   = lane & 15;

  const int bh = blockIdx.y;
  const int b  = bh / NH_, h = bh % NH_;
  const int q0 = blockIdx.x * 128;
  const int qrow = q0 + wave * 16;

  const size_t qbase = (size_t)bh * NQ_ * HD_;
  const size_t kbase = (size_t)bh * KL_ * HD_;

  AFrag qf[4];
  {
    int r = qrow + l16;
    const unsigned short* qp = Qb + qbase + (size_t)r * HD_;
    #pragma unroll
    for (int kk = 0; kk < 4; ++kk) {
      int ko = kk * 32 + lhalf * 8;
      qf[kk].u[0] = *(const uint4*)(qp + ko);
      qf[kk].u[1] = *(const uint4*)(qp + ko + 16);
    }
  }

  f32x8 oacc[8];
  #pragma unroll
  for (int t = 0; t < 8; ++t)
    #pragma unroll
    for (int e = 0; e < 8; ++e) oacc[t][e] = 0.f;
  float mrow[8], lrow[8];
  #pragma unroll
  for (int r = 0; r < 8; ++r) { mrow[r] = -1e30f; lrow[r] = 0.f; }

  for (int kc = 0; kc < KL_; kc += 32) {
    { // stage K (key, hd) directly
      int row = tid >> 3;
      int col = (tid & 7) * 16;
      const unsigned short* src = Kb + kbase + (size_t)(kc + row) * HD_ + col;
      *(uint4*)&Ks[row][col]     = *(const uint4*)(src);
      *(uint4*)&Ks[row][col + 8] = *(const uint4*)(src + 8);
    }
    { // stage V transposed (hd, key)
      int row = tid >> 3;
      int col = (tid & 7) * 16;
      const unsigned short* src = Vb + kbase + (size_t)(kc + row) * HD_ + col;
      unsigned short tmp[16];
      *(uint4*)&tmp[0] = *(const uint4*)(src);
      *(uint4*)&tmp[8] = *(const uint4*)(src + 8);
      #pragma unroll
      for (int e = 0; e < 16; ++e) Vt[col + e][row] = tmp[e];
    }
    __syncthreads();

    // S = Q * K^T over 2 sub-chunks of 16 keys (K-dim = HD, 4 WMMA each)
    f32x8 sacc[2];
    #pragma unroll
    for (int g = 0; g < 2; ++g) {
      #pragma unroll
      for (int e = 0; e < 8; ++e) sacc[g][e] = 0.f;
      #pragma unroll
      for (int kk = 0; kk < 4; ++kk) {
        AFrag bfK;
        int key = g * 16 + l16;
        int ho  = kk * 32 + lhalf * 8;
        bfK.u[0] = *(const uint4*)&Ks[key][ho];
        bfK.u[1] = *(const uint4*)&Ks[key][ho + 16];
        sacc[g] = __builtin_amdgcn_wmma_f32_16x16x32_bf16(
            false, qf[kk].v, false, bfK.v, (short)0, sacc[g], false, false);
      }
    }
    // scale + causal mask (image keys kidx >= NQ always visible)
    #pragma unroll
    for (int g = 0; g < 2; ++g) {
      int kidx = kc + g * 16 + l16;
      #pragma unroll
      for (int r = 0; r < 8; ++r) {
        int q = qrow + lhalf * 8 + r;
        float s = sacc[g][r] * scale;
        if (kidx < NQ_ && kidx > q) s = -1e30f;
        sacc[g][r] = s;
      }
    }
    // online softmax
    float corr[8];
    #pragma unroll
    for (int r = 0; r < 8; ++r) {
      float v = fmaxf(sacc[0][r], sacc[1][r]);
      #pragma unroll
      for (int o = 1; o < 16; o <<= 1) v = fmaxf(v, __shfl_xor(v, o, 32));
      float mnew = fmaxf(mrow[r], v);
      float c  = __expf(mrow[r] - mnew);
      float p0 = __expf(sacc[0][r] - mnew);
      float p1 = __expf(sacc[1][r] - mnew);
      sacc[0][r] = p0; sacc[1][r] = p1;
      float rs = p0 + p1;
      #pragma unroll
      for (int o = 1; o < 16; o <<= 1) rs += __shfl_xor(rs, o, 32);
      lrow[r] = lrow[r] * c + rs;
      mrow[r] = mnew;
      corr[r] = c;
    }
    #pragma unroll
    for (int t = 0; t < 8; ++t)
      #pragma unroll
      for (int r = 0; r < 8; ++r) oacc[t][r] *= corr[r];

    // C-layout -> A-layout via LDS (bf16 P tile)
    #pragma unroll
    for (int g = 0; g < 2; ++g)
      #pragma unroll
      for (int r = 0; r < 8; ++r)
        Ps[wave][lhalf * 8 + r][g * 16 + l16] = f32_to_bf16_bits(sacc[g][r]);
    __syncthreads();

    AFrag pf;
    {
      int row = l16;
      int ko  = lhalf * 8;
      pf.u[0] = *(const uint4*)&Ps[wave][row][ko];
      pf.u[1] = *(const uint4*)&Ps[wave][row][16 + ko];
    }
    #pragma unroll
    for (int t = 0; t < 8; ++t) {
      AFrag bfV;
      int hd = t * 16 + l16;
      int ko = lhalf * 8;
      bfV.u[0] = *(const uint4*)&Vt[hd][ko];
      bfV.u[1] = *(const uint4*)&Vt[hd][ko + 16];
      oacc[t] = __builtin_amdgcn_wmma_f32_16x16x32_bf16(
          false, pf.v, false, bfV.v, (short)0, oacc[t], false, false);
    }
    __syncthreads();
  }

  #pragma unroll
  for (int t = 0; t < 8; ++t)
    #pragma unroll
    for (int r = 0; r < 8; ++r) {
      int q = qrow + lhalf * 8 + r;
      int d = t * 16 + l16;
      float v = oacc[t][r] / lrow[r];
      Ob[((size_t)b * NQ_ + q) * C_ + h * HD_ + d] = f32_to_bf16_bits(v);
    }
}

// =====================================================================
// Helper kernels
// =====================================================================
__global__ void cvt_f32_to_bf16(const float* __restrict__ src,
                                unsigned short* __restrict__ dst, size_t n)
{
  size_t i = (size_t)blockIdx.x * blockDim.x + threadIdx.x;
  if (i < n) dst[i] = f32_to_bf16_bits(src[i]);
}

__global__ __launch_bounds__(256)
void rmsnorm_kernel(const float* __restrict__ x, const float* __restrict__ w,
                    float* __restrict__ outF, unsigned short* __restrict__ outB)
{
  int row = blockIdx.x;
  const float* xr = x + (size_t)row * C_;
  __shared__ float red[8];
  __shared__ float total;
  float s = 0.f;
  for (int c = threadIdx.x; c < C_; c += 256) { float v = xr[c]; s += v * v; }
  #pragma unroll
  for (int o = 16; o > 0; o >>= 1) s += __shfl_xor(s, o, 32);
  if ((threadIdx.x & 31) == 0) red[threadIdx.x >> 5] = s;
  __syncthreads();
  if (threadIdx.x == 0) {
    float t = 0.f;
    for (int i = 0; i < 8; ++i) t += red[i];
    total = rsqrtf(t / (float)C_ + 1e-5f);
  }
  __syncthreads();
  float rs = total;
  for (int c = threadIdx.x; c < C_; c += 256) {
    float v = xr[c] * rs * w[c];
    if (outF) outF[(size_t)row * C_ + c] = v;
    outB[(size_t)row * C_ + c] = f32_to_bf16_bits(v);
  }
}

// rope + pack  src:(B,Lsrc,C) f32 -> dst:(B,NH,Ldst,HD) bf16 at key offset
__global__ void rope_pack(const float* __restrict__ src, unsigned short* __restrict__ dst,
                          int Lsrc, int Ldst, int dstOff, int posBase, int doRope)
{
  size_t i = (size_t)blockIdx.x * blockDim.x + threadIdx.x;
  size_t total = (size_t)B_ * Lsrc * C_;
  if (i >= total) return;
  int d = (int)(i % HD_);
  size_t t = i / HD_;
  int hN = (int)(t % NH_); size_t t2 = t / NH_;
  int l = (int)(t2 % Lsrc); int b = (int)(t2 / Lsrc);
  float x = src[i];
  float v = x;
  if (doRope) {
    int j = d & 63;
    float inv = powf(ROPE_THETA_, -2.0f * (float)j / (float)HD_);
    float ang = (float)(posBase + l) * inv;
    float c, s;
    __sincosf(ang, &s, &c);
    float other = (d < 64) ? -src[i + 64] : src[i - 64];
    v = x * c + other * s;
  }
  dst[(((size_t)b * NH_ + hN) * Ldst + dstOff + l) * HD_ + d] = f32_to_bf16_bits(v);
}

// depthwise conv 3x3 same over LR x LR, input sliced from rmsnormed h
__global__ void conv_dw_kernel(const float* __restrict__ h, const float* __restrict__ w,
                               const float* __restrict__ bias, float* __restrict__ out)
{
  int i = blockIdx.x * blockDim.x + threadIdx.x;
  int total = BG_ * OFFD_ * LR_ * LR_;
  if (i >= total) return;
  int x = i % LR_; int y = (i / LR_) % LR_;
  int c = (i / (LR_ * LR_)) % OFFD_; int bg = i / (LR_ * LR_ * OFFD_);
  int b = bg / OFFG_, g = bg % OFFG_;
  float acc = bias[c];
  #pragma unroll
  for (int ky = 0; ky < 3; ++ky)
    #pragma unroll
    for (int kx = 0; kx < 3; ++kx) {
      int yy = y + ky - 1, xx = x + kx - 1;
      if (yy < 0 || yy >= LR_ || xx < 0 || xx >= LR_) continue;
      float v = h[((size_t)b * NQ_ + (yy * LR_ + xx)) * C_ + g * OFFD_ + c];
      acc += v * w[c * 9 + ky * 3 + kx];
    }
  out[i] = acc;
}

// LayerNorm over channels (per spatial pos) + quickGELU
__global__ __launch_bounds__(256)
void ln2d_gelu_kernel(const float* __restrict__ xin, const float* __restrict__ w,
                      const float* __restrict__ bv, float* __restrict__ out)
{
  int pos = blockIdx.x;                       // bg*576 + yx
  int bg = pos / (LR_ * LR_); int yx = pos % (LR_ * LR_);
  __shared__ float rs[8], rs2[8], mv[2];
  float s = 0.f, s2 = 0.f;
  for (int c = threadIdx.x; c < OFFD_; c += 256) {
    float v = xin[((size_t)bg * OFFD_ + c) * (LR_ * LR_) + yx];
    s += v; s2 += v * v;
  }
  #pragma unroll
  for (int o = 16; o > 0; o >>= 1) { s += __shfl_xor(s, o, 32); s2 += __shfl_xor(s2, o, 32); }
  if ((threadIdx.x & 31) == 0) { rs[threadIdx.x >> 5] = s; rs2[threadIdx.x >> 5] = s2; }
  __syncthreads();
  if (threadIdx.x == 0) {
    float a = 0.f, b2 = 0.f;
    for (int i = 0; i < 8; ++i) { a += rs[i]; b2 += rs2[i]; }
    float m = a / (float)OFFD_;
    float var = b2 / (float)OFFD_ - m * m;
    mv[0] = m; mv[1] = rsqrtf(var + 1e-6f);
  }
  __syncthreads();
  float m = mv[0], inv = mv[1];
  for (int c = threadIdx.x; c < OFFD_; c += 256) {
    size_t idx = ((size_t)bg * OFFD_ + c) * (LR_ * LR_) + yx;
    float v = (xin[idx] - m) * inv * w[c] + bv[c];
    out[idx] = v * (1.f / (1.f + __expf(-1.702f * v)));
  }
}

// x2[bg][i][yx] = sum_c xg[bg][c][yx]*W[c][i] + bias[i]
__global__ void lrproj_kernel(const float* __restrict__ xg, const float* __restrict__ W,
                              const float* __restrict__ bias, float* __restrict__ out)
{
  int idx = blockIdx.x * blockDim.x + threadIdx.x;
  int total = BG_ * INTER_ * LR_ * LR_;
  if (idx >= total) return;
  int yx = idx % (LR_ * LR_); int i = (idx / (LR_ * LR_)) % INTER_;
  int bg = idx / (LR_ * LR_ * INTER_);
  float acc = bias[i];
  const float* xp = xg + (size_t)bg * OFFD_ * (LR_ * LR_) + yx;
  for (int c = 0; c < OFFD_; ++c) acc += xp[(size_t)c * (LR_ * LR_)] * W[c * INTER_ + i];
  out[idx] = acc;
}

__global__ void mean_h_kernel(const float* __restrict__ h, float* __restrict__ out)
{
  int i = blockIdx.x * blockDim.x + threadIdx.x;
  if (i >= B_ * C_) return;
  int b = i / C_, c = i % C_;
  float s = 0.f;
  for (int q = 0; q < NQ_; ++q) s += h[((size_t)b * NQ_ + q) * C_ + c];
  out[i] = s / (float)NQ_;
}

__global__ void intent_kernel(const float* __restrict__ meanh, const float* __restrict__ W,
                              const float* __restrict__ bias, float* __restrict__ out)
{
  int i = blockIdx.x * blockDim.x + threadIdx.x;
  if (i >= B_ * INTER_) return;
  int b = i / INTER_, j = i % INTER_;
  float acc = bias[j];
  for (int c = 0; c < C_; ++c) acc += meanh[b * C_ + c] * W[c * INTER_ + j];
  out[i] = acc;
}

// cat-LN + Woff + tanh offsets + grid; gyx: [gy(BG*576), gx(BG*576)]
__global__ __launch_bounds__(256)
void offsets_kernel(const float* __restrict__ x2, const float* __restrict__ intent,
                    const float* __restrict__ w, const float* __restrict__ bb,
                    const float* __restrict__ Woff, float* __restrict__ gyx)
{
  int pos = blockIdx.x;
  int bg = pos / (LR_ * LR_); int yx = pos % (LR_ * LR_);
  int b = bg / OFFG_;
  __shared__ float rs[8], rs2[8], mv[2], rd0[8], rd1[8];
  auto catv = [&](int c) -> float {
    if (c < INTER_) return x2[((size_t)bg * INTER_ + c) * (LR_ * LR_) + yx];
    return intent[b * INTER_ + (c - INTER_)];
  };
  float s = 0.f, s2 = 0.f;
  for (int c = threadIdx.x; c < 2 * INTER_; c += 256) { float v = catv(c); s += v; s2 += v * v; }
  #pragma unroll
  for (int o = 16; o > 0; o >>= 1) { s += __shfl_xor(s, o, 32); s2 += __shfl_xor(s2, o, 32); }
  if ((threadIdx.x & 31) == 0) { rs[threadIdx.x >> 5] = s; rs2[threadIdx.x >> 5] = s2; }
  __syncthreads();
  if (threadIdx.x == 0) {
    float a = 0.f, b2 = 0.f;
    for (int i = 0; i < 8; ++i) { a += rs[i]; b2 += rs2[i]; }
    float m = a / (float)(2 * INTER_);
    float var = b2 / (float)(2 * INTER_) - m * m;
    mv[0] = m; mv[1] = rsqrtf(var + 1e-6f);
  }
  __syncthreads();
  float m = mv[0], inv = mv[1];
  float d0 = 0.f, d1 = 0.f;
  for (int c = threadIdx.x; c < 2 * INTER_; c += 256) {
    float v = (catv(c) - m) * inv * w[c] + bb[c];
    d0 += v * Woff[c * 2 + 0];
    d1 += v * Woff[c * 2 + 1];
  }
  #pragma unroll
  for (int o = 16; o > 0; o >>= 1) { d0 += __shfl_xor(d0, o, 32); d1 += __shfl_xor(d1, o, 32); }
  if ((threadIdx.x & 31) == 0) { rd0[threadIdx.x >> 5] = d0; rd1[threadIdx.x >> 5] = d1; }
  __syncthreads();
  if (threadIdx.x == 0) {
    float a0 = 0.f, a1 = 0.f;
    for (int i = 0; i < 8; ++i) { a0 += rd0[i]; a1 += rd1[i]; }
    int y = yx / LR_, x = yx % LR_;
    float gy0 = (0.5f + (float)y) / (float)(LR_ - 1) * 2.f - 1.f;
    float gx0 = (0.5f + (float)x) / (float)(LR_ - 1) * 2.f - 1.f;
    float gy = fminf(1.f, fmaxf(-1.f, gy0 + tanhf(a0) * (2.f / (float)LR_)));
    float gx = fminf(1.f, fmaxf(-1.f, gx0 + tanhf(a1) * (2.f / (float)LR_)));
    gyx[bg * (LR_ * LR_) + yx] = gy;
    gyx[BG_ * (LR_ * LR_) + bg * (LR_ * LR_) + yx] = gx;
  }
}

__global__ void bilinear_kernel(const float* __restrict__ hdfeat, const float* __restrict__ gyx,
                                unsigned short* __restrict__ sampB)
{
  size_t i = (size_t)blockIdx.x * blockDim.x + threadIdx.x;
  size_t total = (size_t)B_ * NIMG_ * C_;
  if (i >= total) return;
  int cc = (int)(i % C_); size_t t = i / C_;
  int p = (int)(t % NIMG_); int b = (int)(t / NIMG_);
  int g = cc / OFFD_; int c = cc % OFFD_;
  int bg = b * OFFG_ + g;
  float gy = gyx[bg * NIMG_ + p];
  float gx = gyx[BG_ * NIMG_ + bg * NIMG_ + p];
  float py = (gy + 1.f) * 0.5f * (float)(HR_ - 1);
  float px = (gx + 1.f) * 0.5f * (float)(HR_ - 1);
  float y0f = fminf(fmaxf(floorf(py), 0.f), (float)(HR_ - 1));
  float x0f = fminf(fmaxf(floorf(px), 0.f), (float)(HR_ - 1));
  float y1f = fminf(y0f + 1.f, (float)(HR_ - 1));
  float x1f = fminf(x0f + 1.f, (float)(HR_ - 1));
  float wy = py - y0f, wx = px - x0f;
  int y0 = (int)y0f, y1 = (int)y1f, x0 = (int)x0f, x1 = (int)x1f;
  auto at = [&](int yy, int xx) -> float {
    return hdfeat[((size_t)b * (HR_ * HR_) + yy * HR_ + xx) * C_ + g * OFFD_ + c];
  };
  float top = at(y0, x0) * (1.f - wx) + at(y0, x1) * wx;
  float bot = at(y1, x0) * (1.f - wx) + at(y1, x1) * wx;
  float v = top * (1.f - wy) + bot * wy;
  sampB[i] = f32_to_bf16_bits(v);
}

__global__ void silu_mul_kernel(const float* __restrict__ g, const float* __restrict__ u,
                                unsigned short* __restrict__ out, size_t n)
{
  size_t i = (size_t)blockIdx.x * blockDim.x + threadIdx.x;
  if (i >= n) return;
  float x = g[i];
  float v = (x / (1.f + __expf(-x))) * u[i];
  out[i] = f32_to_bf16_bits(v);
}

// =====================================================================
// Host launch
// =====================================================================
extern "C" void kernel_launch(void* const* d_in, const int* in_sizes, int n_in,
                              void* d_out, int out_size, void* d_ws, size_t ws_size,
                              hipStream_t stream)
{
  const float* hidden   = (const float*)d_in[0];
  const float* hdfeat   = (const float*)d_in[1];
  const float* ln_in_w  = (const float*)d_in[2];
  const float* ln_post_w= (const float*)d_in[3];
  const float* Wq       = (const float*)d_in[4];
  const float* Wk       = (const float*)d_in[5];
  const float* Wv       = (const float*)d_in[6];
  const float* Wo       = (const float*)d_in[7];
  const float* conv_w   = (const float*)d_in[8];
  const float* conv_b   = (const float*)d_in[9];
  const float* ln1_w    = (const float*)d_in[10];
  const float* ln1_b    = (const float*)d_in[11];
  const float* Wlrproj  = (const float*)d_in[12];
  const float* blrproj  = (const float*)d_in[13];
  const float* Wint     = (const float*)d_in[14];
  const float* bint     = (const float*)d_in[15];
  const float* ln2_w    = (const float*)d_in[16];
  const float* ln2_b    = (const float*)d_in[17];
  const float* Woff     = (const float*)d_in[18];
  const float* Wk_hd    = (const float*)d_in[19];
  const float* Wv_hd    = (const float*)d_in[20];
  const float* Wgate    = (const float*)d_in[21];
  const float* Wup      = (const float*)d_in[22];
  const float* Wdown    = (const float*)d_in[23];

  char* ws = (char*)d_ws;
  size_t off = 0;
  auto alloc = [&](size_t bytes) -> void* {
    void* p = ws + off;
    off = (off + bytes + 255) & ~(size_t)255;
    return p;
  };

  const size_t nBNC  = (size_t)B_ * NQ_ * C_;       // 4.19M
  const size_t nIMGC = (size_t)B_ * NIMG_ * C_;     // 2.36M
  const size_t nCC   = (size_t)C_ * C_;
  const size_t nCD   = (size_t)C_ * DFF_;

  float*          h_f32   = (float*)alloc(nBNC * 4);
  unsigned short* h_bf    = (unsigned short*)alloc(nBNC * 2);
  unsigned short* Wq_bf   = (unsigned short*)alloc(nCC * 2);
  unsigned short* Wk_bf   = (unsigned short*)alloc(nCC * 2);
  unsigned short* Wv_bf   = (unsigned short*)alloc(nCC * 2);
  unsigned short* Wo_bf   = (unsigned short*)alloc(nCC * 2);
  unsigned short* Wkhd_bf = (unsigned short*)alloc(nCC * 2);
  unsigned short* Wvhd_bf = (unsigned short*)alloc(nCC * 2);
  unsigned short* Wg_bf   = (unsigned short*)alloc(nCD * 2);
  unsigned short* Wu_bf   = (unsigned short*)alloc(nCD * 2);
  unsigned short* Wd_bf   = (unsigned short*)alloc(nCD * 2);
  float*          q_f32   = (float*)alloc(nBNC * 4);
  float*          k_f32   = (float*)alloc(nBNC * 4);
  float*          v_f32   = (float*)alloc(nBNC * 4);
  float*          khd_f32 = (float*)alloc(nIMGC * 4);
  float*          vhd_f32 = (float*)alloc(nIMGC * 4);
  unsigned short* Qb      = (unsigned short*)alloc((size_t)B_ * NH_ * NQ_ * HD_ * 2);
  unsigned short* Kb      = (unsigned short*)alloc((size_t)B_ * NH_ * KL_ * HD_ * 2);
  unsigned short* Vb      = (unsigned short*)alloc((size_t)B_ * NH_ * KL_ * HD_ * 2);
  float*          convo   = (float*)alloc((size_t)BG_ * OFFD_ * NIMG_ * 4);
  float*          xg      = (float*)alloc((size_t)BG_ * OFFD_ * NIMG_ * 4);
  float*          x2      = (float*)alloc((size_t)BG_ * INTER_ * NIMG_ * 4);
  float*          meanh   = (float*)alloc((size_t)B_ * C_ * 4);
  float*          intent  = (float*)alloc((size_t)B_ * INTER_ * 4);
  float*          gyx     = (float*)alloc((size_t)2 * BG_ * NIMG_ * 4);
  unsigned short* samp_bf = (unsigned short*)alloc(nIMGC * 2);
  unsigned short* attno   = (unsigned short*)alloc(nBNC * 2);
  float*          h2_f32  = (float*)alloc(nBNC * 4);
  unsigned short* m_bf    = (unsigned short*)alloc(nBNC * 2);
  float*          gate_f  = (float*)alloc((size_t)B_ * NQ_ * DFF_ * 4);
  float*          up_f    = (float*)alloc((size_t)B_ * NQ_ * DFF_ * 4);
  unsigned short* act_bf  = (unsigned short*)alloc((size_t)B_ * NQ_ * DFF_ * 2);

  auto cvt = [&](const float* s, unsigned short* d, size_t n) {
    unsigned blocks = (unsigned)((n + 255) / 256);
    cvt_f32_to_bf16<<<blocks, 256, 0, stream>>>(s, d, n);
  };
  cvt(Wq, Wq_bf, nCC);   cvt(Wk, Wk_bf, nCC);   cvt(Wv, Wv_bf, nCC);
  cvt(Wo, Wo_bf, nCC);   cvt(Wk_hd, Wkhd_bf, nCC); cvt(Wv_hd, Wvhd_bf, nCC);
  cvt(Wgate, Wg_bf, nCD); cvt(Wup, Wu_bf, nCD);  cvt(Wdown, Wd_bf, nCD);

  // h = rmsnorm(hidden) * ln_in_w
  rmsnorm_kernel<<<B_ * NQ_, 256, 0, stream>>>(hidden, ln_in_w, h_f32, h_bf);

  auto gemm = [&](const unsigned short* A, const unsigned short* Bmx,
                  const float* bias, const float* resid,
                  float* oF, unsigned short* oB, int M, int N, int K) {
    dim3 g((unsigned)(N / TN), (unsigned)(M / TM));
    gemm_bf16_wmma<<<g, 256, 0, stream>>>(A, Bmx, bias, resid, oF, oB, M, N, K);
  };

  // QKV projections
  gemm(h_bf, Wq_bf, nullptr, nullptr, q_f32, nullptr, B_ * NQ_, C_, C_);
  gemm(h_bf, Wk_bf, nullptr, nullptr, k_f32, nullptr, B_ * NQ_, C_, C_);
  gemm(h_bf, Wv_bf, nullptr, nullptr, v_f32, nullptr, B_ * NQ_, C_, C_);

  // deformable-offset path
  {
    int n1 = BG_ * OFFD_ * NIMG_;
    conv_dw_kernel<<<(n1 + 255) / 256, 256, 0, stream>>>(h_f32, conv_w, conv_b, convo);
    ln2d_gelu_kernel<<<BG_ * NIMG_, 256, 0, stream>>>(convo, ln1_w, ln1_b, xg);
    int n2 = BG_ * INTER_ * NIMG_;
    lrproj_kernel<<<(n2 + 255) / 256, 256, 0, stream>>>(xg, Wlrproj, blrproj, x2);
    mean_h_kernel<<<(B_ * C_ + 255) / 256, 256, 0, stream>>>(h_f32, meanh);
    intent_kernel<<<(B_ * INTER_ + 255) / 256, 256, 0, stream>>>(meanh, Wint, bint, intent);
    offsets_kernel<<<BG_ * NIMG_, 256, 0, stream>>>(x2, intent, ln2_w, ln2_b, Woff, gyx);
    size_t n3 = (size_t)B_ * NIMG_ * C_;
    bilinear_kernel<<<(unsigned)((n3 + 255) / 256), 256, 0, stream>>>(hdfeat, gyx, samp_bf);
  }

  // sampled-image K/V projections  (M = B*NIMG = 1152 = 9*128)
  gemm(samp_bf, Wkhd_bf, nullptr, nullptr, khd_f32, nullptr, B_ * NIMG_, C_, C_);
  gemm(samp_bf, Wvhd_bf, nullptr, nullptr, vhd_f32, nullptr, B_ * NIMG_, C_, C_);

  // rope + pack into (B, NH, L, HD) bf16
  auto rp = [&](const float* s, unsigned short* d, int Lsrc, int Ldst, int dOff,
                int posBase, int doRope) {
    size_t n = (size_t)B_ * Lsrc * C_;
    rope_pack<<<(unsigned)((n + 255) / 256), 256, 0, stream>>>(s, d, Lsrc, Ldst, dOff, posBase, doRope);
  };
  rp(q_f32,   Qb, NQ_,   NQ_, 0,    0, 1);
  rp(k_f32,   Kb, NQ_,   KL_, 0,    0, 1);
  rp(khd_f32, Kb, NIMG_, KL_, NQ_,  0, 1);   // IMG_START = 0
  rp(v_f32,   Vb, NQ_,   KL_, 0,    0, 0);
  rp(vhd_f32, Vb, NIMG_, KL_, NQ_,  0, 0);

  // flash attention
  {
    dim3 g((unsigned)(NQ_ / 128), (unsigned)(B_ * NH_));
    attn_flash_wmma<<<g, 256, 0, stream>>>(Qb, Kb, Vb, attno, 1.0f / sqrtf((float)HD_));
  }

  // output projection + residual -> h2
  gemm(attno, Wo_bf, nullptr, hidden, h2_f32, nullptr, B_ * NQ_, C_, C_);

  // post-norm
  rmsnorm_kernel<<<B_ * NQ_, 256, 0, stream>>>(h2_f32, ln_post_w, nullptr, m_bf);

  // MLP
  gemm(m_bf, Wg_bf, nullptr, nullptr, gate_f, nullptr, B_ * NQ_, DFF_, C_);
  gemm(m_bf, Wu_bf, nullptr, nullptr, up_f,   nullptr, B_ * NQ_, DFF_, C_);
  {
    size_t n = (size_t)B_ * NQ_ * DFF_;
    silu_mul_kernel<<<(unsigned)((n + 255) / 256), 256, 0, stream>>>(gate_f, up_f, act_bf, n);
  }
  gemm(act_bf, Wd_bf, nullptr, h2_f32, (float*)d_out, nullptr, B_ * NQ_, C_, DFF_);

  (void)in_sizes; (void)n_in; (void)out_size; (void)ws_size;
}
